// DynamicRNNV2_87522843560199
// MI455X (gfx1250) — matrix operand order repaired and checked by reading, not compiled, not run on hardware
//
#include <hip/hip_runtime.h>
#include <hip/hip_bf16.h>

// GRU (Keras reset_after=True): B=16, T=4096, D=512, H=512
//   Phase 1: xk = x @ W + b0      (big bf16 WMMA GEMM, async global->LDS staging)
//   Phase 2: sequential recurrence, 16-WG persistent cooperative kernel
//            (async LDS staging of h, cluster-barrier + atomic grid sync)

#define B_SZ  16
#define T_LEN 4096
#define D_IN  512
#define H_DIM 512
#define N3H   1536
#define M_ROWS (B_SZ * T_LEN)   // 65536

#define AS1 __attribute__((address_space(1)))
#define AS3 __attribute__((address_space(3)))

typedef __attribute__((ext_vector_type(16))) __bf16 v16bf;
typedef __attribute__((ext_vector_type(8)))  float  v8f;
typedef __attribute__((__vector_size__(4 * sizeof(int)))) int v4i;

union Frag {            // one WMMA 16-bit A/B operand: 16 bf16 = 32 bytes
    v16bf v;
    uint4 q[2];
};

#if __has_builtin(__builtin_amdgcn_global_load_async_to_lds_b128)
#define HAVE_ASYNC_LDS 1
static __device__ __forceinline__ void async_copy_b128(const void* gsrc, void* lds) {
    __builtin_amdgcn_global_load_async_to_lds_b128(
        (AS1 v4i*)gsrc, (AS3 v4i*)lds, 0, 0);
}
static __device__ __forceinline__ void wait_async0() {
#if __has_builtin(__builtin_amdgcn_s_wait_asynccnt)
    __builtin_amdgcn_s_wait_asynccnt(0);
#else
    asm volatile("s_wait_asynccnt 0x0" ::: "memory");
#endif
}
#endif

static __device__ __forceinline__ __bf16 f2bf(float f) {
    // round-to-nearest-even f32 -> bf16
    unsigned u = __float_as_uint(f);
    u = (u + 0x7FFFu + ((u >> 16) & 1u)) >> 16;
    unsigned short s = (unsigned short)u;
    __bf16 b;
    __builtin_memcpy(&b, &s, sizeof(b));
    return b;
}

// ---------------------------------------------------------------- phase 0
__global__ void cvt_f32_bf16(const float* __restrict__ in,
                             __bf16* __restrict__ out, int n) {
    int i = blockIdx.x * blockDim.x + threadIdx.x;
    int stride = gridDim.x * blockDim.x;
    for (; i < n; i += stride) out[i] = f2bf(in[i]);
}

// in: [K=512][N=1536] row-major  ->  out: [N=1536][K=512] bf16 (transposed)
__global__ void transpose_cvt(const float* __restrict__ in,
                              __bf16* __restrict__ out) {
    int i = blockIdx.x * blockDim.x + threadIdx.x;
    int stride = gridDim.x * blockDim.x;
    for (; i < D_IN * N3H; i += stride) {
        int n = i >> 9;         // / 512
        int k = i & 511;
        out[i] = f2bf(in[k * N3H + n]);
    }
}

__global__ void zero_state(__bf16* __restrict__ hbuf0,
                           unsigned* __restrict__ sync_ctr) {
    int i = blockIdx.x * blockDim.x + threadIdx.x;
    if (i < B_SZ * H_DIM) hbuf0[i] = f2bf(0.0f);
    if (i == 0) *sync_ctr = 0u;
}

// ---------------------------------------------------------------- phase 1
// xk[M][1536] = X[M][512] (bf16) @ Wt^T + bias0    (Wt stored [N][K])
// 256 threads = 8 waves; block tile 128x128; wave tile 64x32 (4x2 WMMA tiles)
__global__ __launch_bounds__(256) void gemm_xk(
    const __bf16* __restrict__ X,     // [M][512]
    const __bf16* __restrict__ Wt,    // [1536][512]
    const float*  __restrict__ bias0, // [1536]
    float* __restrict__ XK)           // [M][1536]
{
    __shared__ __bf16 As[128][32];    // [m][k] tile of X
    __shared__ __bf16 Bs[128][32];    // [n][k] tile of Wt (already K-major)

    const int tid  = threadIdx.x;
    const int lane = tid & 31;
    const int wave = tid >> 5;
    const int wm   = wave >> 2;       // 0..1  (64-row slab)
    const int wn   = wave & 3;        // 0..3  (32-col slab)
    const int m0   = blockIdx.x * 128;
    const int n0   = blockIdx.y * 128;

    v8f acc[4][2];
    for (int i = 0; i < 4; ++i)
        for (int j = 0; j < 2; ++j)
            for (int vi = 0; vi < 8; ++vi) acc[i][j][vi] = 0.0f;

    const int lrow   = tid >> 1;           // 0..127
    const int lchunk = (tid & 1) << 4;     // 0 or 16 (bf16 elements)

    const int arow   = (lane & 15);
    const int khalfA = (lane >> 4) << 3;   // 0 or 8   (A K-half per ISA layout)
    const int kh16B  = (lane >> 4) << 4;   // 0 or 16  (unused; B loads use khalfA, same k order as A)

    for (int k0 = 0; k0 < 512; k0 += 32) {
        // cooperative tile load: each thread moves 16 bf16 (2 x b128) per tile
        const __bf16* gA = &X[(size_t)(m0 + lrow) * 512 + k0 + lchunk];
        const __bf16* gB = &Wt[(size_t)(n0 + lrow) * 512 + k0 + lchunk];
#ifdef HAVE_ASYNC_LDS
        // async global->LDS (ASYNCcnt-tracked, no VGPR staging)
        async_copy_b128(gA,     &As[lrow][lchunk]);
        async_copy_b128(gA + 8, &As[lrow][lchunk + 8]);
        async_copy_b128(gB,     &Bs[lrow][lchunk]);
        async_copy_b128(gB + 8, &Bs[lrow][lchunk + 8]);
        wait_async0();
#else
        {
            const uint4* a4 = (const uint4*)gA;
            uint4 a0 = a4[0], a1 = a4[1];
            *(uint4*)&As[lrow][lchunk]     = a0;
            *(uint4*)&As[lrow][lchunk + 8] = a1;
            const uint4* b4 = (const uint4*)gB;
            uint4 b0 = b4[0], b1 = b4[1];
            *(uint4*)&Bs[lrow][lchunk]     = b0;
            *(uint4*)&Bs[lrow][lchunk + 8] = b1;
        }
#endif
        __syncthreads();

        Frag afr[4], bfr[2];
        for (int i = 0; i < 4; ++i) {
            const __bf16* p = &As[wm * 64 + i * 16 + arow][0];
            afr[i].q[0] = *(const uint4*)&p[khalfA];
            afr[i].q[1] = *(const uint4*)&p[16 + khalfA];
        }
        for (int j = 0; j < 2; ++j) {
            const __bf16* p = &Bs[wn * 32 + j * 16 + (lane & 15)][0];
            bfr[j].q[0] = *(const uint4*)&p[khalfA];
            bfr[j].q[1] = *(const uint4*)&p[16 + khalfA];
        }
        for (int i = 0; i < 4; ++i)
            for (int j = 0; j < 2; ++j)
                acc[i][j] = __builtin_amdgcn_wmma_f32_16x16x32_bf16(
                    false, afr[i].v, false, bfr[j].v,
                    (short)0, acc[i][j], false, false);
        __syncthreads();
    }

    // epilogue: + bias0, store f32
    for (int i = 0; i < 4; ++i) {
        const int rbase = m0 + wm * 64 + i * 16 + ((lane >> 4) << 3);
        for (int j = 0; j < 2; ++j) {
            const int col = n0 + wn * 32 + j * 16 + (lane & 15);
            const float bv = bias0[col];
            for (int vi = 0; vi < 8; ++vi)
                XK[(size_t)(rbase + vi) * N3H + col] = acc[i][j][vi] + bv;
        }
    }
}

// ---------------------------------------------------------------- phase 2
// 16 workgroups (== max cluster size), 192 threads = 6 waves each.
// WG b owns h columns [32b, 32b+32); wave w computes gate g=w/2, col-half w%2.
__global__ __launch_bounds__(192) void gru_rec(
    const float*  __restrict__ XK,        // [M][1536]
    const __bf16* __restrict__ Ut,        // [1536][512]  (K-major rows)
    const float*  __restrict__ bias1,     // [1536] recurrent bias
    __bf16*       __restrict__ hping,     // [2][16][512] bf16 ping-pong
    unsigned*     __restrict__ sync_ctr,
    float*        __restrict__ out,       // [16][4096][512]
    float*        __restrict__ out_state) // [16][512]
{
    __shared__ __bf16 hbS[16][512];       // staged h (16 KB), shared by all 6 waves
    __shared__ float recS[3][16][32];     // z/r/h partial recurrences
    __shared__ float hS[16][32];          // this WG's f32 h slice

    const int tid  = threadIdx.x;
    const int lane = tid & 31;
    const int wave = tid >> 5;            // 0..5
    const int g    = wave >> 1;           // gate: 0=z 1=r 2=h
    const int sub  = wave & 1;            // 16-col half
    const int j0   = blockIdx.x * 32;
    const int nWG  = gridDim.x;           // 16

    for (int idx = tid; idx < 512; idx += 192)
        hS[idx >> 5][idx & 31] = 0.0f;
    __syncthreads();

    const int   ncol   = g * 512 + j0 + sub * 16 + (lane & 15);
    const __bf16* urow = &Ut[(size_t)ncol * 512];
    const float biasc  = bias1[ncol];
    const int arow     = lane & 15;            // batch row
    const int khalfA   = (lane >> 4) << 3;
    const int kh16B    = (lane >> 4) << 4;

    for (int t = 0; t < T_LEN; ++t) {
        const __bf16* hb = hping + (size_t)(t & 1) * (16 * 512);

        // stage h[16][512] bf16 into LDS once per step (1024 x b128 chunks)
        for (int idx = tid; idx < 1024; idx += 192) {
#ifdef HAVE_ASYNC_LDS
            async_copy_b128(&hb[idx * 8], &hbS[0][0] + idx * 8);
#else
            *(uint4*)(&hbS[0][0] + idx * 8) = *(const uint4*)&hb[idx * 8];
#endif
        }
#ifdef HAVE_ASYNC_LDS
        wait_async0();
#endif
        __syncthreads();

        v8f acc;
        for (int vi = 0; vi < 8; ++vi) acc[vi] = biasc;  // C init = rec bias (col-uniform)

        for (int k0 = 0; k0 < 512; k0 += 32) {
            Frag a, b;
            const __bf16* hp = &hbS[arow][k0];
            a.q[0] = *(const uint4*)&hp[khalfA];
            a.q[1] = *(const uint4*)&hp[16 + khalfA];
            b.q[0] = *(const uint4*)&urow[k0 + khalfA];
            b.q[1] = *(const uint4*)&urow[k0 + 16 + khalfA];
            acc = __builtin_amdgcn_wmma_f32_16x16x32_bf16(
                false, a.v, false, b.v, (short)0, acc, false, false);
        }

        // scatter rec tile to LDS: rows m = (lane/16)*8 + vi, col = sub*16 + lane%16
        {
            const int j  = sub * 16 + (lane & 15);
            const int mb = (lane >> 4) << 3;
            for (int vi = 0; vi < 8; ++vi) recS[g][mb + vi][j] = acc[vi];
        }
        __syncthreads();

        // gate elementwise update for this WG's 16x32 slice
        __bf16* hnext = hping + (size_t)((t + 1) & 1) * (16 * 512);
        for (int idx = tid; idx < 512; idx += 192) {
            const int m = idx >> 5, j = idx & 31;
            const size_t row = (size_t)m * T_LEN + t;
            const float* xp = &XK[row * N3H + j0 + j];
            const float xz = xp[0], xr = xp[512], xh = xp[1024];
            const float rz = recS[0][m][j];
            const float rr = recS[1][m][j];
            const float rh = recS[2][m][j];
            const float z  = 1.0f / (1.0f + __expf(-(xz + rz)));
            const float r  = 1.0f / (1.0f + __expf(-(xr + rr)));
            const float hh = tanhf(xh + r * rh);
            const float hp = hS[m][j];
            const float hn = z * hp + (1.0f - z) * hh;
            hS[m][j] = hn;
            out[row * H_DIM + j0 + j] = hn;
            hnext[m * 512 + j0 + j]   = f2bf(hn);
            if (t == T_LEN - 1) out_state[m * 512 + j0 + j] = hn;
            // warm next step's xk lines (global_prefetch_b8)
            if (t + 1 < T_LEN)
                __builtin_prefetch(&XK[((size_t)m * T_LEN + t + 1) * N3H + j0 + j], 0, 1);
        }
        __syncthreads();

        // ---- grid-wide step barrier ----
        if (tid == 0) {
            __builtin_amdgcn_fence(__ATOMIC_RELEASE, "agent");
            __hip_atomic_fetch_add(sync_ctr, 1u, __ATOMIC_RELAXED,
                                   __HIP_MEMORY_SCOPE_AGENT);
            const unsigned target = (unsigned)nWG * (unsigned)(t + 1);
            while (__hip_atomic_load(sync_ctr, __ATOMIC_RELAXED,
                                     __HIP_MEMORY_SCOPE_AGENT) < target) {
                __builtin_amdgcn_s_sleep(1);
            }
            __builtin_amdgcn_fence(__ATOMIC_ACQUIRE, "agent");
        }
        // cluster split-barrier (s_barrier_signal/wait -3); NOP if not clustered
        __builtin_amdgcn_s_cluster_barrier();
        __syncthreads();
    }
}

// ---------------------------------------------------------------- launch
extern "C" void kernel_launch(void* const* d_in, const int* in_sizes, int n_in,
                              void* d_out, int out_size, void* d_ws, size_t ws_size,
                              hipStream_t stream) {
    (void)in_sizes; (void)n_in; (void)out_size; (void)ws_size;

    const float* x    = (const float*)d_in[0];   // [16][4096][512]
    const float* W    = (const float*)d_in[1];   // [512][1536]
    const float* U    = (const float*)d_in[2];   // [512][1536]
    const float* bias = (const float*)d_in[3];   // [2][1536]

    char* ws = (char*)d_ws;
    size_t off = 0;
    __bf16* Xb    = (__bf16*)(ws + off); off += (size_t)M_ROWS * 512 * 2;    // 64 MB
    __bf16* Wt    = (__bf16*)(ws + off); off += (size_t)N3H * 512 * 2;       // 1.5 MB
    __bf16* Ut    = (__bf16*)(ws + off); off += (size_t)N3H * 512 * 2;       // 1.5 MB
    float*  XK    = (float*)(ws + off);  off += (size_t)M_ROWS * N3H * 4;    // 384 MB
    __bf16* hping = (__bf16*)(ws + off); off += (size_t)2 * 16 * 512 * 2;
    unsigned* sync_ctr = (unsigned*)(ws + off); off += 256;

    float* out       = (float*)d_out;
    float* out_state = out + (size_t)B_SZ * T_LEN * H_DIM;

    cvt_f32_bf16<<<2048, 256, 0, stream>>>(x, Xb, M_ROWS * 512);
    transpose_cvt<<<768, 256, 0, stream>>>(W, Wt);
    transpose_cvt<<<768, 256, 0, stream>>>(U, Ut);
    zero_state<<<(B_SZ * H_DIM + 255) / 256, 256, 0, stream>>>(hping, sync_ctr);

    dim3 g1(M_ROWS / 128, N3H / 128);   // 512 x 12 workgroups
    gemm_xk<<<g1, 256, 0, stream>>>(Xb, Wt, bias, XK);

    gru_rec<<<16, 192, 0, stream>>>(XK, Ut, bias + N3H, hping, sync_ctr,
                                    out, out_state);
}
